// BiLSTM_CRF_31619549233734
// MI455X (gfx1250) — matrix operand order repaired and checked by reading, not compiled, not run on hardware
//
#include <hip/hip_runtime.h>
#include <hip/hip_bf16.h>
#include <math.h>

typedef _Float16 f16;
typedef __attribute__((ext_vector_type(16))) _Float16 v16h;
typedef __attribute__((ext_vector_type(8)))  _Float16 v8h;
typedef __attribute__((ext_vector_type(8)))  float    v8f;

#define B_   64
#define T_   256
#define E_   256
#define HD_  256
#define NT_  20
#define G4_  1024   /* 4*HD */

// ---------------------------------------------------------------------------
// WMMA helpers (CDNA5 16x16x32 f16 -> f32), per CDNA5 ISA layouts.
// ---------------------------------------------------------------------------
__device__ __forceinline__ v8f wmma_f16(v16h a, v16h b, v8f c) {
  return __builtin_amdgcn_wmma_f32_16x16x32_f16(
      /*neg_a=*/false, a, /*neg_b=*/false, b,
      /*c_mod=*/(short)0, c, /*reuse_a=*/false, /*reuse_b=*/false);
}

// A tile (16x32, MxK) from row-major X (lda halves). Lane l<16: row M=l,
// elems 0..7 = K0..7, 8..15 = K16..23; lanes 16..31: K shifted by 8.
__device__ __forceinline__ v16h load_a_rm(const f16* X, int lda, int m0, int k0, int lane) {
  int m = m0 + (lane & 15);
  int koff = (lane >> 4) * 8;
  const f16* p = X + (size_t)m * lda + k0 + koff;
  v8h lo = *(const v8h*)p;
  v8h hi = *(const v8h*)(p + 16);
  v16h a;
#pragma unroll
  for (int j = 0; j < 8; ++j) { a[j] = lo[j]; a[8 + j] = hi[j]; }
  return a;
}

// B tile (32x16, KxN) where B[k][n] = W[n][k], W row-major (N x K, ldw halves).
// Lane: n = lane&15; elems 0..7 = K0..7, 8..15 = K16..23; lanes 16..31: K shifted by 8.
__device__ __forceinline__ v16h load_b_wt(const f16* W, int ldw, int n0, int k0, int lane) {
  int n = n0 + (lane & 15);
  int koff = (lane >> 4) * 8;
  const f16* p = W + (size_t)n * ldw + k0 + koff;
  v8h lo = *(const v8h*)p;
  v8h hi = *(const v8h*)(p + 16);
  v16h b;
#pragma unroll
  for (int j = 0; j < 8; ++j) { b[j] = lo[j]; b[8 + j] = hi[j]; }
  return b;
}

// ---------------------------------------------------------------------------
// Small utility kernels
// ---------------------------------------------------------------------------
__global__ void cast_f32_f16(const float* __restrict__ s, f16* __restrict__ d, int n) {
  int i = blockIdx.x * blockDim.x + threadIdx.x;
  if (i < n) d[i] = (f16)s[i];
}

__global__ void zero_u32(unsigned* __restrict__ p, int n) {
  int i = blockIdx.x * blockDim.x + threadIdx.x;
  if (i < n) p[i] = 0u;
}

__global__ void init_out(float* o) { o[0] = 0.f; }

// Embedding gather -> f16 (stride E_), one block per (b,t)
__global__ void embed_kernel(const int* __restrict__ tokens,
                             const float* __restrict__ emb,
                             f16* __restrict__ X) {
  int row = blockIdx.x;
  int tok = tokens[row];
  X[(size_t)row * E_ + threadIdx.x] = (f16)emb[(size_t)tok * E_ + threadIdx.x];
}

// ---------------------------------------------------------------------------
// gx = X @ W_ih^T + (b_ih + b_hh), stored f16.
// Register-blocked: each wave computes 1 M-tile x 4 N-tiles, so each A
// fragment is reused by 4 WMMAs. grid = (M/16, 2, 2 dirs), block = 256.
// ---------------------------------------------------------------------------
__global__ void gemm_gx(const f16* __restrict__ X, int K,
                        const f16* __restrict__ Wih,
                        const float* __restrict__ bih, const float* __restrict__ bhh,
                        f16* __restrict__ GX) {
  int lane = threadIdx.x & 31;
  int wave = threadIdx.x >> 5;
  int mt = blockIdx.x;                   // 0..1023 (M/16)
  int ng = blockIdx.y * 8 + wave;        // 0..15 (groups of 4 N-tiles)
  int dir = blockIdx.z;
  const f16* W = Wih + (size_t)dir * G4_ * K;
  f16* out = GX + (size_t)dir * (B_ * T_) * G4_;
  int m0 = mt * 16, n0 = ng * 64;
  v8f acc[4];
#pragma unroll
  for (int q = 0; q < 4; ++q) acc[q] = (v8f){};
  for (int k0 = 0; k0 < K; k0 += 32) {
    if (k0 + 32 < K)
      __builtin_prefetch(W + (size_t)(n0 + (lane & 15)) * K + k0 + 32, 0, 1);
    v16h a = load_a_rm(X, K, m0, k0, lane);
#pragma unroll
    for (int q = 0; q < 4; ++q) {
      v16h b = load_b_wt(W, K, n0 + q * 16, k0, lane);
      acc[q] = wmma_f16(a, b, acc[q]);
    }
  }
  int mb = m0 + (lane >> 4) * 8;
#pragma unroll
  for (int q = 0; q < 4; ++q) {
    int n = n0 + q * 16 + (lane & 15);
    float bias = bih[dir * G4_ + n] + bhh[dir * G4_ + n];
#pragma unroll
    for (int r = 0; r < 8; ++r)
      out[(size_t)(mb + r) * G4_ + n] = (f16)(acc[q][r] + bias);
  }
}

// ---------------------------------------------------------------------------
// Persistent LSTM recurrence. grid = 2 (dir), block = 1024 (32 waves).
// LDS: h (64x256 f16, 32KB) + g (64x1024 f32, 256KB) + lens (256B) = ~288KB
// of the WGP's 320KB. Per step: g = h@Whh^T (WMMA, A from LDS, B from L2),
// with each W_hh fragment loaded ONCE and reused across all 4 M-tiles
// (4 accumulators) -- minimizes off-chip traffic on the serial critical path.
// Then gates in registers (c state: 16 floats/thread).
// OUT = X16 buffer, stride 2*HD, col offset dir*HD; masked, bwd un-reversed.
// ---------------------------------------------------------------------------
__global__ void __launch_bounds__(1024)
lstm_scan(const f16* __restrict__ GXall, const f16* __restrict__ Whh,
          const float* __restrict__ bih, const float* __restrict__ bhh,
          const int* __restrict__ lens, f16* __restrict__ OUT) {
  extern __shared__ char smem[];
  f16*   sh_h   = (f16*)smem;                                // 64*256 f16
  float* sh_g   = (float*)(smem + B_ * HD_ * 2);             // 64*1024 f32
  int*   sh_len = (int*)(smem + B_ * HD_ * 2 + B_ * G4_ * 4);

  int tid = threadIdx.x;
  int lane = tid & 31, wave = tid >> 5;
  int dir = blockIdx.x;
  const f16* gx = GXall + (size_t)dir * (B_ * T_) * G4_;
  const f16* W  = Whh + (size_t)dir * G4_ * HD_;

  for (int i = tid; i < B_ * HD_; i += 1024) sh_h[i] = (f16)0.f;
  if (tid < B_) sh_len[tid] = lens[tid];
  __syncthreads();

  // gate-phase ownership: thread owns batch b=tid>>4, 16 consecutive hd.
  int bown = tid >> 4;
  int hd0 = (tid & 15) * 16;
  float c[16];
#pragma unroll
  for (int j = 0; j < 16; ++j) c[j] = 0.f;

  int n0base = wave * 32;  // this wave's two 16-wide N tiles

  for (int t = 0; t < T_; ++t) {
    // ---- phase 1: g = h @ Whh^T + gx ----
    for (int ntl = 0; ntl < 2; ++ntl) {
      int nn0 = n0base + ntl * 16;
      int ncol = nn0 + (lane & 15);
      float bias2 = bih[dir * G4_ + ncol] + bhh[dir * G4_ + ncol];
      v8f acc[4];
#pragma unroll
      for (int q = 0; q < 4; ++q) acc[q] = (v8f){};
#pragma unroll
      for (int k0 = 0; k0 < HD_; k0 += 32) {
        v16h bm = load_b_wt(W, HD_, nn0, k0, lane);   // fetched once...
#pragma unroll
        for (int mt2 = 0; mt2 < 4; ++mt2) {           // ...reused 4x
          int m = mt2 * 16 + (lane & 15);
          int koff = (lane >> 4) * 8;
          const f16* p = sh_h + m * HD_ + k0 + koff;  // ds_read_b128 x2
          v8h lo = *(const v8h*)p;
          v8h hi = *(const v8h*)(p + 16);
          v16h a;
#pragma unroll
          for (int j = 0; j < 8; ++j) { a[j] = lo[j]; a[8 + j] = hi[j]; }
          acc[mt2] = wmma_f16(a, bm, acc[mt2]);
        }
      }
#pragma unroll
      for (int mt2 = 0; mt2 < 4; ++mt2) {
        int mb = mt2 * 16 + (lane >> 4) * 8;
#pragma unroll
        for (int r = 0; r < 8; ++r) {
          int b = mb + r;
          int len = sh_len[b];
          float gxv;
          if (dir == 0) {
            gxv = (float)gx[((size_t)b * T_ + t) * G4_ + ncol];
          } else {
            if (t < len) gxv = (float)gx[((size_t)b * T_ + (len - 1 - t)) * G4_ + ncol];
            else         gxv = bias2;   // x_rev == 0 past len -> bias only
          }
          sh_g[b * G4_ + ncol] = acc[mt2][r] + gxv;
        }
      }
    }
    __syncthreads();
    // ---- phase 2: gates, update c/h, masked store ----
    {
      int b = bown;
      int len = sh_len[b];
      bool st = (t < len);
      int trow = st ? ((dir == 0) ? t : (len - 1 - t)) : 0;
      f16* orow = OUT + ((size_t)b * T_ + trow) * (2 * HD_) + dir * HD_;
#pragma unroll
      for (int j = 0; j < 16; ++j) {
        int hd = hd0 + j;
        float gi = sh_g[b * G4_ + hd];
        float gf = sh_g[b * G4_ + HD_ + hd];
        float gc = sh_g[b * G4_ + 2 * HD_ + hd];
        float go = sh_g[b * G4_ + 3 * HD_ + hd];
        float si = 1.f / (1.f + __expf(-gi));
        float sf = 1.f / (1.f + __expf(-gf));
        float so = 1.f / (1.f + __expf(-go));
        float cc = sf * c[j] + si * tanhf(gc);
        c[j] = cc;
        float h = so * tanhf(cc);
        sh_h[b * HD_ + hd] = (f16)h;
        if (st) orow[hd] = (f16)h;
      }
    }
    __syncthreads();
  }
}

// ---------------------------------------------------------------------------
// logits = H @ lin_w^T + lin_b, N padded 20->32 (pad rows zeroed).
// grid = 128, block = 256 (8 waves), wave -> one M tile, both N tiles.
// ---------------------------------------------------------------------------
__global__ void linear_wmma(const f16* __restrict__ H, const f16* __restrict__ Wl,
                            const float* __restrict__ bl, float* __restrict__ logits) {
  int lane = threadIdx.x & 31;
  int wave = threadIdx.x >> 5;
  int mt = blockIdx.x * 8 + wave;  // 0..1023
  int m0 = mt * 16;
  v8f acc0 = {}, acc1 = {};
#pragma unroll
  for (int k0 = 0; k0 < 2 * HD_; k0 += 32) {
    v16h a  = load_a_rm(H, 2 * HD_, m0, k0, lane);
    v16h b0 = load_b_wt(Wl, 2 * HD_, 0, k0, lane);
    v16h b1 = load_b_wt(Wl, 2 * HD_, 16, k0, lane);
    acc0 = wmma_f16(a, b0, acc0);
    acc1 = wmma_f16(a, b1, acc1);
  }
  int mb = m0 + (lane >> 4) * 8;
  int n = lane & 15;
  float bb0 = bl[n];
#pragma unroll
  for (int r = 0; r < 8; ++r)
    logits[(size_t)(mb + r) * NT_ + n] = acc0[r] + bb0;
  if (n < NT_ - 16) {
    float bb1 = bl[16 + n];
#pragma unroll
    for (int r = 0; r < 8; ++r)
      logits[(size_t)(mb + r) * NT_ + 16 + n] = acc1[r] + bb1;
  }
}

// ---------------------------------------------------------------------------
// CRF NLL: one wave per batch; alpha[NT] one state per lane (wave32),
// logsumexp via lane shuffles. atomicAdd of (partition - numerator).
// ---------------------------------------------------------------------------
__device__ __forceinline__ float wred_sum(float v) {
#pragma unroll
  for (int off = 16; off; off >>= 1) v += __shfl_xor(v, off, 32);
  return v;
}
__device__ __forceinline__ float wred_max(float v) {
#pragma unroll
  for (int off = 16; off; off >>= 1) v = fmaxf(v, __shfl_xor(v, off, 32));
  return v;
}

__global__ void crf_nll(const float* __restrict__ logits, const int* __restrict__ labels,
                        const int* __restrict__ lens, const float* __restrict__ trans,
                        const float* __restrict__ start_t, const float* __restrict__ end_t,
                        float* out) {
  int b = blockIdx.x;
  int lane = threadIdx.x;
  int len = lens[b];
  const float* L = logits + (size_t)b * T_ * NT_;
  const int* tg = labels + (size_t)b * T_;
  bool valid = lane < NT_;
  int jj = valid ? lane : 0;
  float alpha = valid ? (start_t[jj] + L[jj]) : -1e30f;

  for (int t = 1; t < len; ++t) {
    float vv[NT_];
    float mx = -1e30f;
#pragma unroll
    for (int i = 0; i < NT_; ++i) {
      float ai = __shfl(alpha, i, 32);
      float v = ai + trans[i * NT_ + jj];
      vv[i] = v;
      mx = fmaxf(mx, v);
    }
    float s = 0.f;
#pragma unroll
    for (int i = 0; i < NT_; ++i) s += __expf(vv[i] - mx);
    float nw = mx + __logf(s) + L[t * NT_ + jj];
    alpha = valid ? nw : -1e30f;
  }

  // numerator pieces, distributed over lanes
  float part = 0.f;
  for (int t = lane; t < len; t += 32) part += L[t * NT_ + tg[t]];
  for (int t = 1 + lane; t < len; t += 32) part += trans[tg[t - 1] * NT_ + tg[t]];
  part = wred_sum(part);

  float v = valid ? (alpha + end_t[jj]) : -1e30f;
  float mx = wred_max(v);
  float s = wred_sum(valid ? __expf(v - mx) : 0.f);
  if (lane == 0) {
    float numerator = part + start_t[tg[0]] + end_t[tg[len - 1]];
    float partition = mx + __logf(s);
    atomicAdd(out, partition - numerator);
  }
}

// ---------------------------------------------------------------------------
extern "C" void kernel_launch(void* const* d_in, const int* in_sizes, int n_in,
                              void* d_out, int out_size, void* d_ws, size_t ws_size,
                              hipStream_t stream) {
  (void)in_sizes; (void)n_in; (void)out_size; (void)ws_size;
  const int*   tokens  = (const int*)  d_in[0];
  const int*   lens    = (const int*)  d_in[1];
  const int*   labels  = (const int*)  d_in[2];
  const float* emb     = (const float*)d_in[4];
  const float* w_ih0   = (const float*)d_in[5];
  const float* w_hh0   = (const float*)d_in[6];
  const float* b_ih0   = (const float*)d_in[7];
  const float* b_hh0   = (const float*)d_in[8];
  const float* w_ih1   = (const float*)d_in[9];
  const float* w_hh1   = (const float*)d_in[10];
  const float* b_ih1   = (const float*)d_in[11];
  const float* b_hh1   = (const float*)d_in[12];
  const float* lin_w   = (const float*)d_in[13];
  const float* lin_b   = (const float*)d_in[14];
  const float* trans   = (const float*)d_in[15];
  const float* start_t = (const float*)d_in[16];
  const float* end_t   = (const float*)d_in[17];

  char* ws = (char*)d_ws;
  size_t off = 0;
  auto take = [&](size_t bytes) {
    char* p = ws + off;
    off += (bytes + 255) & ~(size_t)255;
    return p;
  };
  f16*   X16   = (f16*)take((size_t)B_ * T_ * 512 * 2);   // emb / h0 / h1 (16MB)
  f16*   Wih0  = (f16*)take((size_t)2 * 1024 * 256 * 2);
  f16*   Whh0  = (f16*)take((size_t)2 * 1024 * 256 * 2);
  f16*   Wih1  = (f16*)take((size_t)2 * 1024 * 512 * 2);
  f16*   Whh1  = (f16*)take((size_t)2 * 1024 * 256 * 2);
  f16*   Lin16 = (f16*)take((size_t)32 * 512 * 2);        // N padded to 32
  f16*   GX    = (f16*)take((size_t)2 * B_ * T_ * G4_ * 2); // 67MB, reused
  float* LOG   = (float*)take((size_t)B_ * T_ * NT_ * 4);

  auto blocks = [](int n) { return (unsigned)((n + 255) / 256); };

  // weights -> f16 (padded linear weight zeroed first)
  cast_f32_f16<<<blocks(2 * 1024 * 256), 256, 0, stream>>>(w_ih0, Wih0, 2 * 1024 * 256);
  cast_f32_f16<<<blocks(2 * 1024 * 256), 256, 0, stream>>>(w_hh0, Whh0, 2 * 1024 * 256);
  cast_f32_f16<<<blocks(2 * 1024 * 512), 256, 0, stream>>>(w_ih1, Wih1, 2 * 1024 * 512);
  cast_f32_f16<<<blocks(2 * 1024 * 256), 256, 0, stream>>>(w_hh1, Whh1, 2 * 1024 * 256);
  zero_u32<<<blocks(32 * 512 * 2 / 4), 256, 0, stream>>>((unsigned*)Lin16, 32 * 512 * 2 / 4);
  cast_f32_f16<<<blocks(NT_ * 512), 256, 0, stream>>>(lin_w, Lin16, NT_ * 512);

  // embedding gather (f16, stride E_)
  embed_kernel<<<dim3(B_ * T_), 256, 0, stream>>>(tokens, emb, X16);

  int x16_u32 = (int)((size_t)B_ * T_ * 512 * 2 / 4);
  size_t shbytes = (size_t)B_ * HD_ * 2 + (size_t)B_ * G4_ * 4 + 256;

  // ---- layer 0 ----
  gemm_gx<<<dim3(1024, 2, 2), 256, 0, stream>>>(X16, 256, Wih0, b_ih0, b_hh0, GX);
  zero_u32<<<blocks(x16_u32), 256, 0, stream>>>((unsigned*)X16, x16_u32);
  lstm_scan<<<dim3(2), 1024, shbytes, stream>>>(GX, Whh0, b_ih0, b_hh0, lens, X16);

  // ---- layer 1 ----
  gemm_gx<<<dim3(1024, 2, 2), 256, 0, stream>>>(X16, 512, Wih1, b_ih1, b_hh1, GX);
  zero_u32<<<blocks(x16_u32), 256, 0, stream>>>((unsigned*)X16, x16_u32);
  lstm_scan<<<dim3(2), 1024, shbytes, stream>>>(GX, Whh1, b_ih1, b_hh1, lens, X16);

  // ---- head + CRF ----
  linear_wmma<<<dim3(128), 256, 0, stream>>>(X16, Lin16, lin_b, LOG);
  init_out<<<1, 1, 0, stream>>>((float*)d_out);
  crf_nll<<<dim3(B_), 32, 0, stream>>>(LOG, labels, lens, trans, start_t, end_t, (float*)d_out);
}
